// SelfAttention_91010357002971
// MI455X (gfx1250) — compile-verified, matrix operand order checked by reading, not run on hardware
//
#include <hip/hip_runtime.h>
#include <hip/hip_bf16.h>
#include <stdint.h>

// ---------------------------------------------------------------------------
// Bottleneck self-attention for MI455X (gfx1250), wave32 + WMMA bf16.
//   X: [4,256,64,64] fp32.  n = 4096 pixels, d = 128 bottleneck channels.
//   Attention equivalent: Q:=k (index j), K:=q (index i), V:=v; softmax over i.
//   All GEMMs on v_wmma_f32_16x16x32_bf16 (fp32 accum); single-pass flash
//   attention; q/v tiles staged in LDS via GLOBAL_LOAD_ASYNC_TO_LDS_B128,
//   double-buffered on ASYNCcnt, shared by the 8 waves of each workgroup.
// ---------------------------------------------------------------------------

#define IN_CH 256
#define BOT   128
#define NB    4
#define NPIX  4096   // 64*64
#define NIT   (NPIX / 32)

#define QROWP 136    // 128 + 8 bf16 pad  -> 272B pitch (16B aligned, bank-spread)
#define VROWP 40     //  32 + 8 bf16 pad  ->  80B pitch (16B aligned, bank-spread)

typedef __attribute__((ext_vector_type(16))) __bf16 v16bf;
typedef __attribute__((ext_vector_type(8)))  float  v8f;

union Frag16 {
  v16bf v;
  __bf16 h[16];
  uint4  u4[2];
};
union Frag8 {
  __bf16 h[8];
  uint4  u4;
};

// 16-bit A/B fragment K-mapping (ISA 7.12.2): lane holds two contiguous runs
// of 8 K values at K-offsets 8*hi and 16+8*hi  (hi = lane>=16).
__device__ inline v16bf load_frag_bf16(const __bf16* rowbase, int hi) {
  Frag16 f;
  f.u4[0] = *(const uint4*)(rowbase + 8 * hi);
  f.u4[1] = *(const uint4*)(rowbase + 16 + 8 * hi);
  return f.v;
}

// Same mapping but source row is fp32 (weights) -> convert to bf16.
__device__ inline v16bf load_frag_f32row(const float* rowbase, int hi) {
  Frag16 f;
  const float* p0 = rowbase + 8 * hi;
  const float* p1 = rowbase + 16 + 8 * hi;
#pragma unroll
  for (int t = 0; t < 8; ++t) f.h[t] = (__bf16)p0[t];
#pragma unroll
  for (int t = 0; t < 8; ++t) f.h[8 + t] = (__bf16)p1[t];
  return f.v;
}

__device__ inline v8f wmma_bf16(v16bf a, v16bf b, v8f c) {
  return __builtin_amdgcn_wmma_f32_16x16x32_bf16(
      /*neg_a=*/false, a, /*neg_b=*/false, b,
      /*c_mod=*/(short)0, c, /*reuse_a=*/false, /*reuse_b=*/false);
}

// Async copy 16B/lane from global to LDS (ASYNCcnt-tracked).  VDST VGPR holds
// the workgroup-relative LDS byte address (= low 32 bits of a generic __shared__
// pointer), VADDR holds the 64-bit global address.  ISA 15.18.3 opcode 98.
__device__ inline void async_b128(uint32_t lds_off, uint64_t gaddr) {
  asm volatile("global_load_async_to_lds_b128 %0, %1, off"
               :
               : "v"(lds_off), "v"(gaddr)
               : "memory");
}
#define WAIT_ASYNC(n) asm volatile("s_wait_asynccnt " #n ::: "memory")

// ---------------------------------------------------------------------------
// Kernel 1: q/k/v projections.  One wave = one 16x16 (o,n) tile.
//   blockIdx.y = matrix (0=q,1=k,2=v), blockIdx.z = batch.
//   q,k stored transposed [n][c] (contiguous in c for attention fragments);
//   v stored [c][n] (contiguous in i for the PV A-fragment / LDS tile copy).
// ---------------------------------------------------------------------------
__global__ __launch_bounds__(256)
void attn_qkv_proj(const float* __restrict__ X,
                   const float* __restrict__ Wq, const float* __restrict__ bq,
                   const float* __restrict__ Wk, const float* __restrict__ bk,
                   const float* __restrict__ Wv, const float* __restrict__ bv,
                   __bf16* __restrict__ qT, __bf16* __restrict__ kT,
                   __bf16* __restrict__ vN) {
  const int lane = threadIdx.x & 31;
  const int wave = threadIdx.x >> 5;
  const int mode = blockIdx.y;           // 0=q, 1=k, 2=v
  const int b    = blockIdx.z;
  const int tid  = blockIdx.x * 8 + wave;  // 0..2047
  const int o0   = (tid >> 8) * 16;        // 0..112
  const int n0   = (tid & 255) * 16;       // 0..4080
  const int ln   = lane & 15;
  const int hi   = lane >> 4;

  const float* Wm = (mode == 0) ? Wq : (mode == 1 ? Wk : Wv);
  const float* bm = (mode == 0) ? bq : (mode == 1 ? bk : bv);

  v8f acc = {0.f, 0.f, 0.f, 0.f, 0.f, 0.f, 0.f, 0.f};

  const float* Xbase = X + (size_t)b * IN_CH * NPIX + n0 + ln;  // column n0+ln
#pragma unroll 1
  for (int c0 = 0; c0 < IN_CH; c0 += 32) {
    // A = weight tile (M=o, K=c): row-major, contiguous in c.
    v16bf a = load_frag_f32row(Wm + (size_t)(o0 + ln) * IN_CH + c0, hi);
    // B = X tile (K=c, N=n): c strided by NPIX -> gathers (coalesced over n).
    Frag16 fb;
#pragma unroll
    for (int h = 0; h < 16; ++h) {
      int K = h + ((h >= 8) ? 8 : 0) + 8 * hi;
      fb.h[h] = (__bf16)Xbase[(size_t)(c0 + K) * NPIX];
    }
    acc = wmma_bf16(a, fb.v, acc);
  }
  // bias: D-layout rows o = o0 + r + 8*hi
#pragma unroll
  for (int r = 0; r < 8; ++r) acc[r] += bm[o0 + r + 8 * hi];

  if (mode < 2) {
    __bf16* dst = (mode == 0) ? qT : kT;   // transposed [b][n][c]
    Frag8 s;
#pragma unroll
    for (int r = 0; r < 8; ++r) s.h[r] = (__bf16)acc[r];
    *(uint4*)(dst + ((size_t)b * NPIX + n0 + ln) * BOT + o0 + 8 * hi) = s.u4;
  } else {                                  // v: [b][c][n]
#pragma unroll
    for (int r = 0; r < 8; ++r)
      vN[((size_t)b * BOT + o0 + r + 8 * hi) * NPIX + n0 + ln] = (__bf16)acc[r];
  }
}

// ---------------------------------------------------------------------------
// Kernel 2: flash attention.  8 waves/block, wave owns 16 output columns j.
// Per 32-row i-block the workgroup async-copies one 8KB q-tile (contiguous)
// and one 8KB v-tile (128 rows x 64B) into double-buffered LDS (4 async
// instructions per wave per buffer), then every wave runs 8 logit WMMAs +
// online softmax + 8 PV WMMAs off LDS.  S(D-layout)->P(B-layout) is
// lane-local; softmax reductions need one shfl_xor(16) each.
// ---------------------------------------------------------------------------
__global__ __launch_bounds__(256)
void attn_flash(const __bf16* __restrict__ qT, const __bf16* __restrict__ kT,
                const __bf16* __restrict__ vN, __bf16* __restrict__ aoT) {
  __shared__ __align__(16) __bf16 smQ[2][32][QROWP];   // i-local x c (padded)
  __shared__ __align__(16) __bf16 smV[2][BOT][VROWP];  // c x i-local (padded)

  const int tid  = threadIdx.x;
  const int lane = tid & 31;
  const int wave = tid >> 5;
  const int b    = blockIdx.y;
  const int j0   = (blockIdx.x * 8 + wave) * 16;  // 0..4080
  const int ln   = lane & 15;
  const int hi   = lane >> 4;

  const __bf16* qbase = qT + (size_t)b * NPIX * BOT;  // rows of 256B, contiguous
  const __bf16* vbase = vN + (size_t)b * BOT * NPIX;  // rows stride 8192B

  // Issue the async tile copy for i-block i0 into buffer `buf`.
  // 512 chunks of 16B each for Q and for V; thread handles chunks tid, tid+256.
  auto issue = [&](int i0, int buf) {
    const char*    qg   = (const char*)(qbase + (size_t)i0 * BOT);
    const char*    vg   = (const char*)(vbase + i0);
    const uint32_t qlds = (uint32_t)(uintptr_t)&smQ[buf][0][0];
    const uint32_t vlds = (uint32_t)(uintptr_t)&smV[buf][0][0];
#pragma unroll
    for (int s = 0; s < 512; s += 256) {
      const int ch = tid + s;
      // Q tile is contiguous in global: byte ch*16 -> padded row layout in LDS.
      async_b128(qlds + (uint32_t)((ch >> 4) * (QROWP * 2) + (ch & 15) * 16),
                 (uint64_t)(uintptr_t)(qg + (size_t)ch * 16));
      // V tile: row = ch>>2 (c), 4 chunks of 16B per 64B row.
      async_b128(vlds + (uint32_t)((ch >> 2) * (VROWP * 2) + (ch & 3) * 16),
                 (uint64_t)(uintptr_t)(vg + (size_t)(ch >> 2) * 8192 +
                                       (size_t)(ch & 3) * 16));
    }
  };

  // Hoist the 16 "query" columns (= k_j) for the whole i loop: 4 fragments.
  const __bf16* krow = kT + ((size_t)b * NPIX + j0 + ln) * BOT;
  v16bf kf[4];
#pragma unroll
  for (int cc = 0; cc < 4; ++cc) kf[cc] = load_frag_bf16(krow + cc * 32, hi);

  v8f acc[8];
#pragma unroll
  for (int t = 0; t < 8; ++t)
#pragma unroll
    for (int r = 0; r < 8; ++r) acc[t][r] = 0.f;

  float m = -__builtin_inff();
  float l = 0.f;

  issue(0, 0);  // prologue: fill buffer 0

#pragma unroll 1
  for (int it = 0; it < NIT; ++it) {
    const int buf = it & 1;
    if (it + 1 < NIT) {
      issue((it + 1) * 32, buf ^ 1);  // overlap next tile with this compute
      WAIT_ASYNC(4);                  // in-order: oldest 4 (= this buf) done
    } else {
      WAIT_ASYNC(0);
    }
    __syncthreads();                  // all waves' copies visible in LDS

    // Logits: S0 rows i0..i0+15, S1 rows i0+16..i0+31  (A = q^T, K=c=128)
    v8f S0 = {0.f, 0.f, 0.f, 0.f, 0.f, 0.f, 0.f, 0.f};
    v8f S1 = {0.f, 0.f, 0.f, 0.f, 0.f, 0.f, 0.f, 0.f};
#pragma unroll
    for (int cc = 0; cc < 4; ++cc) {
      v16bf a0 = load_frag_bf16(&smQ[buf][ln][cc * 32], hi);
      v16bf a1 = load_frag_bf16(&smQ[buf][16 + ln][cc * 32], hi);
      S0 = wmma_bf16(a0, kf[cc], S0);
      S1 = wmma_bf16(a1, kf[cc], S1);
    }

    // Online softmax over i, per column j.  Lane pair (L, L+16) shares column.
    float mloc = -__builtin_inff();
#pragma unroll
    for (int r = 0; r < 8; ++r) mloc = fmaxf(mloc, fmaxf(S0[r], S1[r]));
    mloc = fmaxf(mloc, __shfl_xor(mloc, 16, 32));
    const float mnew  = fmaxf(m, mloc);
    const float scale = __expf(m - mnew);  // first iter: exp(-inf)=0
    float p0[8], p1[8], ls = 0.f;
#pragma unroll
    for (int r = 0; r < 8; ++r) {
      p0[r] = __expf(S0[r] - mnew);
      p1[r] = __expf(S1[r] - mnew);
      ls += p0[r] + p1[r];
    }
    ls += __shfl_xor(ls, 16, 32);
    l = l * scale + ls;
    m = mnew;
#pragma unroll
    for (int t = 0; t < 8; ++t)
#pragma unroll
      for (int r = 0; r < 8; ++r) acc[t][r] *= scale;

    // P as B-fragment (K=i over 32 rows): purely lane-local re-pack.
    Frag16 pf;
#pragma unroll
    for (int r = 0; r < 8; ++r) {
      pf.h[r]     = (__bf16)p0[r];
      pf.h[8 + r] = (__bf16)p1[r];
    }
    // O += V * P   (A = v tile, M=c, K=i=32; 8 c-tiles cover 128 channels)
#pragma unroll
    for (int t = 0; t < 8; ++t) {
      v16bf vf = load_frag_bf16(&smV[buf][16 * t + ln][0], hi);
      acc[t] = wmma_bf16(vf, pf.v, acc[t]);
    }

    __syncthreads();  // everyone done reading `buf` before it is refilled
  }

  // Normalize and store transposed [b][n][c] (contiguous in c for kernel 3).
  const float rl = 1.f / l;
  __bf16* arow = aoT + ((size_t)b * NPIX + j0 + ln) * BOT;
#pragma unroll
  for (int t = 0; t < 8; ++t) {
    Frag8 s;
#pragma unroll
    for (int r = 0; r < 8; ++r) s.h[r] = (__bf16)(acc[t][r] * rl);
    *(uint4*)(arow + 16 * t + 8 * hi) = s.u4;
  }
}

// ---------------------------------------------------------------------------
// Kernel 3: output projection + residual.  Y = gamma*(Wo@attn + bo) + X.
// One wave = one 16x16 (o,n) tile, K=c=128 -> 4 WMMAs.
// ---------------------------------------------------------------------------
__global__ __launch_bounds__(256)
void attn_out_proj(const float* __restrict__ X, const float* __restrict__ Wo,
                   const float* __restrict__ bo, const float* __restrict__ gamma,
                   const __bf16* __restrict__ aoT, float* __restrict__ Y) {
  const int lane = threadIdx.x & 31;
  const int wave = threadIdx.x >> 5;
  const int b    = blockIdx.y;
  const int tid  = blockIdx.x * 8 + wave;  // 0..4095
  const int o0   = (tid >> 8) * 16;        // 0..240
  const int n0   = (tid & 255) * 16;
  const int ln   = lane & 15;
  const int hi   = lane >> 4;

  v8f acc = {0.f, 0.f, 0.f, 0.f, 0.f, 0.f, 0.f, 0.f};
  const float*  wrow = Wo + (size_t)(o0 + ln) * BOT;          // [256][128]
  const __bf16* arow = aoT + ((size_t)b * NPIX + n0 + ln) * BOT;
#pragma unroll
  for (int cc = 0; cc < 4; ++cc) {
    v16bf a  = load_frag_f32row(wrow + cc * 32, hi);
    v16bf bf = load_frag_bf16(arow + cc * 32, hi);
    acc = wmma_bf16(a, bf, acc);
  }

  const float g = gamma[0];
  const int   n = n0 + ln;
#pragma unroll
  for (int r = 0; r < 8; ++r) {
    const int o = o0 + r + 8 * hi;
    const size_t idx = ((size_t)b * IN_CH + o) * NPIX + n;
    Y[idx] = g * (acc[r] + bo[o]) + X[idx];
  }
}

// ---------------------------------------------------------------------------
extern "C" void kernel_launch(void* const* d_in, const int* in_sizes, int n_in,
                              void* d_out, int out_size, void* d_ws, size_t ws_size,
                              hipStream_t stream) {
  const float* X     = (const float*)d_in[0];
  const float* Wq    = (const float*)d_in[1];
  const float* bq    = (const float*)d_in[2];
  const float* Wk    = (const float*)d_in[3];
  const float* bk    = (const float*)d_in[4];
  const float* Wv    = (const float*)d_in[5];
  const float* bv    = (const float*)d_in[6];
  const float* Wo    = (const float*)d_in[7];
  const float* bo    = (const float*)d_in[8];
  const float* gamma = (const float*)d_in[9];
  float* Y = (float*)d_out;

  // Workspace: qT, kT (transposed [b][n][c]), v ([b][c][n]), attn-out^T.
  // 4 x (4*4096*128) bf16 = 16 MiB total (fits easily in 192 MB L2).
  const size_t elems = (size_t)NB * NPIX * BOT;
  __bf16* qT  = (__bf16*)d_ws;
  __bf16* kT  = qT + elems;
  __bf16* vN  = kT + elems;
  __bf16* aoT = vN + elems;

  attn_qkv_proj<<<dim3(256, 3, NB), 256, 0, stream>>>(X, Wq, bq, Wk, bk, Wv, bv,
                                                      qT, kT, vN);
  attn_flash<<<dim3(32, NB), 256, 0, stream>>>(qT, kT, vN, aoT);
  attn_out_proj<<<dim3(512, NB), 256, 0, stream>>>(X, Wo, bo, gamma, aoT, Y);
}
